// BiGCN_12824772345980
// MI455X (gfx1250) — compile-verified, matrix operand order checked by reading, not run on hardware
//
#include <hip/hip_runtime.h>
#include <hip/hip_bf16.h>

// ---------------------------------------------------------------------------
// BiGCN forward for MI455X (gfx1250, wave32, WMMA).
// Dense GEMMs run on v_wmma_f32_16x16x32_bf16 (f32 accumulate); edge
// aggregation uses global_atomic_add_f32 (L2-resident working set).
// Input pointer order assumes setup_inputs() dict insertion order with the
// params dict flattened in its own insertion order.
// ---------------------------------------------------------------------------

#define N_NODES 65536
#define B_GRAPH 128
#define E_EDGES 524288
#define F_IN    256
#define F_HID   128
#define F_OUT   128
#define N_CLASS 4
#define MAXHOP  10
#define NPG     512

typedef __attribute__((ext_vector_type(16))) __bf16 v16bf;
typedef __attribute__((ext_vector_type(8)))  __bf16 v8bf;
typedef __attribute__((ext_vector_type(8)))  float  v8f;

// --------------------------- small utility kernels --------------------------

__global__ void fill_zero_kernel(float* p, int n) {
  int t = blockIdx.x * blockDim.x + threadIdx.x;
  if (t < n) p[t] = 0.0f;
}

__global__ void cvt_bf16_kernel(__bf16* dst, const float* src, int n) {
  int t = blockIdx.x * blockDim.x + threadIdx.x;
  if (t < n) dst[t] = (__bf16)src[t];
}

// w [K,128] f32 row-major -> wT [128,K] bf16 row-major
__global__ void wcvt_t_kernel(__bf16* wT, const float* w, int K) {
  int t = blockIdx.x * blockDim.x + threadIdx.x;
  if (t < K * 128) {
    int k = t >> 7, n = t & 127;
    wT[(size_t)n * K + k] = (__bf16)w[t];
  }
}

__global__ void deg_count_kernel(const int* src, const int* dst,
                                 float* deg_td, float* deg_bu, int E) {
  int e = blockIdx.x * blockDim.x + threadIdx.x;
  if (e < E) {
    atomicAdd(&deg_td[dst[e]], 1.0f);   // TD: target degree by dst
    atomicAdd(&deg_bu[src[e]], 1.0f);   // BU: edges reversed -> target = src
  }
}

// deg -> 1/sqrt(deg + 1 self loop), in place, both arrays
__global__ void deg_to_dinv_kernel(float* a, float* b, int n) {
  int t = blockIdx.x * blockDim.x + threadIdx.x;
  if (t < n) {
    a[t] = rsqrtf(a[t] + 1.0f);
    b[t] = rsqrtf(b[t] + 1.0f);
  }
}

// ------------------------------ WMMA GEMM -----------------------------------
// C[M,128] = A[M,K](bf16 row-major) x B[K,128] given as BT[128,K](bf16).
// 8 waves/block; wave w computes rows [blk*128 + 16w .. +15], all 128 cols.
__global__ __launch_bounds__(256)
void gemm_bf16_wmma_kernel(float* __restrict__ C,
                           const __bf16* __restrict__ A,
                           const __bf16* __restrict__ BT,
                           int K) {
  const int lane = threadIdx.x & 31;
  const int wave = threadIdx.x >> 5;
  const int r    = lane & 15;   // M (A) / N (B) index within 16-tile
  const int kh   = lane >> 4;   // K-half select per ISA 16-bit layout
  const long m0  = (long)blockIdx.x * 128 + wave * 16;

  const __bf16* arow = A + (m0 + r) * (long)K + kh * 8;

  v8f acc[8];
#pragma unroll
  for (int nt = 0; nt < 8; ++nt)
#pragma unroll
    for (int i = 0; i < 8; ++i) acc[nt][i] = 0.0f;

  for (int k0 = 0; k0 < K; k0 += 32) {
    __builtin_prefetch(arow + k0 + 32, 0, 1);
    // A fragment: lane covers K = {k0+kh*8..+7} and {k0+16+kh*8..+7}
    v8bf alo = *(const v8bf*)(arow + k0);
    v8bf ahi = *(const v8bf*)(arow + k0 + 16);
    v16bf a;
#pragma unroll
    for (int i = 0; i < 8; ++i) { a[i] = alo[i]; a[8 + i] = ahi[i]; }

#pragma unroll
    for (int nt = 0; nt < 8; ++nt) {
      const __bf16* brow = BT + (long)(nt * 16 + r) * K + k0 + kh * 8;
      v8bf blo = *(const v8bf*)(brow);
      v8bf bhi = *(const v8bf*)(brow + 16);
      v16bf bfr;
#pragma unroll
      for (int i = 0; i < 8; ++i) { bfr[i] = blo[i]; bfr[8 + i] = bhi[i]; }
      acc[nt] = __builtin_amdgcn_wmma_f32_16x16x32_bf16(
          false, a, false, bfr, (short)0, acc[nt], false, false);
    }
  }

  // D layout: vgpr r8, lane l -> M = r8 + 8*(l>>4), N = (l&15)
#pragma unroll
  for (int nt = 0; nt < 8; ++nt)
#pragma unroll
    for (int r8 = 0; r8 < 8; ++r8)
      C[(m0 + r8 + 8 * kh) * 128 + nt * 16 + r] = acc[nt][r8];
}

// --------------------------- GCN aggregation --------------------------------

// agg[i][c] = dinv[i]^2 * hw[i][c] + bias[c]  (self loop + bias fold)
__global__ void agg_init_kernel(float* agg, const float* hw,
                                const float* dinv, const float* bias, int n) {
  int t = blockIdx.x * blockDim.x + threadIdx.x;
  if (t < n) {
    int i = t >> 7, c = t & 127;
    float di = dinv[i];
    agg[t] = di * di * hw[t] + bias[c];
  }
}

// one thread per (edge, 4 channels): agg[d] += dinv[s]*dinv[d]*hw[s]
__global__ void edge_scatter_kernel(float* agg, const float* hw,
                                    const int* s, const int* d,
                                    const float* dinv, int E) {
  int t = blockIdx.x * blockDim.x + threadIdx.x;
  if (t >= E * 32) return;
  int e  = t >> 5;
  int c4 = (t & 31) << 2;
  int si = s[e], di = d[e];
  float nrm = dinv[si] * dinv[di];
  const float4 v = *(const float4*)(hw + (size_t)si * 128 + c4);
  float* dp = agg + (size_t)di * 128 + c4;
  atomicAdd(dp + 0, nrm * v.x);
  atomicAdd(dp + 1, nrm * v.y);
  atomicAdd(dp + 2, nrm * v.z);
  atomicAdd(dp + 3, nrm * v.w);
}

// cat[i][0:128]=relu(x2[i]), cat[i][128:384]=relu(x[root(i)]) -> bf16
__global__ void cat_relu_kernel(__bf16* cat, const float* x2,
                                const float* x, int n) {
  int t = blockIdx.x * blockDim.x + threadIdx.x;
  if (t < n) {
    int i = t / 384, c = t - i * 384;
    float v;
    if (c < 128) v = x2[(size_t)i * 128 + c];
    else         v = x[(size_t)((i >> 9) << 9) * 256 + (c - 128)];
    cat[t] = (__bf16)fmaxf(v, 0.0f);
  }
}

// per graph: out[b][c]   = mean_j relu(agg2[b*512+j][c]) + xg[b][c]
//            out[b][128+c]= x2[root_b][c] + xg[b][128+c]
__global__ __launch_bounds__(128)
void pool_kernel(float* out, const float* agg2, const float* x2,
                 const float* xg) {
  int b = blockIdx.x, c = threadIdx.x;
  const float* base = agg2 + (size_t)b * NPG * 128 + c;
  float s = 0.0f;
  for (int j = 0; j < NPG; ++j) s += fmaxf(base[(size_t)j * 128], 0.0f);
  out[b * 256 + c]       = s * (1.0f / (float)NPG) + xg[b * 256 + c];
  out[b * 256 + 128 + c] = x2[(size_t)b * NPG * 128 + c] + xg[b * 256 + 128 + c];
}

// ----------------------- propagation chain (tiny) ---------------------------

__device__ inline float block_reduce_sum(float v, float* red, int t) {
  red[t] = v; __syncthreads();
  for (int off = 64; off > 0; off >>= 1) {
    if (t < off) red[t] += red[t + off];
    __syncthreads();
  }
  float r = red[0]; __syncthreads();
  return r;
}

__global__ __launch_bounds__(128)
void prop_kernel(const float* user_state,
                 const float* wu0_w, const float* wu0_b,
                 const float* ws0_b, const float* wd0_b,
                 const float* wu_w, const float* wu_b,
                 const float* ws_w, const float* ws_b,
                 const float* wd_w, const float* wd_b,
                 const float* wx_w, const float* wx_b,
                 const float* lu_w, const float* lu_b,
                 const float* ls_w, const float* ls_b,
                 const float* ld_w, const float* ld_b,
                 const float* raw_alpha, const float* raw_beta,
                 const int* num_hop,
                 float* xg, float* out_u, float* out_s, float* out_d) {
  __shared__ float sU[128], sS[128], sD[128], red[128], cat[384];
  const int b = blockIdx.x, t = threadIdx.x;

  float p = 0.0f;
  for (int i = t; i < 600; i += 128) p += user_state[b * 600 + i];
  float u0 = block_reduce_sum(p, red, t);

  float alpha = 1.0f / (1.0f + expf(-raw_alpha[0]));
  float beta  = 1.0f / (1.0f + expf(-raw_beta[0]));
  float g = 1.0f - alpha - beta;

  sU[t] = u0 * wu0_w[t] + wu0_b[t];
  sS[t] = ws0_b[t];
  sD[t] = wd0_b[t];
  int hop = num_hop[b] - 1;
  float um = 0.0f, sm = 0.0f, dm = 0.0f;
  __syncthreads();

  for (int h = 0; h < MAXHOP; ++h) {
    float nu = wu_b[t];
    for (int k = 0; k < 128; ++k) nu += g * sU[k] * wu_w[k * 128 + t];
    __syncthreads();
    sU[t] = nu;
    __syncthreads();
    float ns = ws_b[t], nd = wd_b[t];
    for (int k = 0; k < 128; ++k) {
      float uk = sU[k];
      ns += (sS[k] + alpha * uk) * ws_w[k * 128 + t];
      nd += (sD[k] + beta  * uk) * wd_w[k * 128 + t];
    }
    __syncthreads();
    sS[t] = ns; sD[t] = nd;
    if (h == hop) { um = nu; sm = ns; dm = nd; }

    float ru = block_reduce_sum(nu * lu_w[t], red, t);
    if (t == 0) out_u[b * MAXHOP + h] = ru + lu_b[0];
    float rs = block_reduce_sum(ns * ls_w[t], red, t);
    if (t == 0) out_s[b * MAXHOP + h] = rs + ls_b[0];
    float rd = block_reduce_sum(nd * ld_w[t], red, t);
    if (t == 0) out_d[b * MAXHOP + h] = rd + ld_b[0];
  }

  cat[t] = um; cat[128 + t] = sm; cat[256 + t] = dm;
  __syncthreads();
  for (int jj = 0; jj < 2; ++jj) {
    int j = t + jj * 128;
    float a = wx_b[j];
    for (int k = 0; k < 384; ++k) a += cat[k] * wx_w[k * 256 + j];
    xg[b * 256 + j] = a;
  }
}

// ------------------------------ final FC ------------------------------------

__global__ __launch_bounds__(128)
void final_kernel(float* logits, const float* bu, const float* td,
                  const float* fc_w, const float* fc_b) {
  int b = threadIdx.x;
  float z0 = fc_b[0], z1 = fc_b[1], z2 = fc_b[2], z3 = fc_b[3];
  for (int c = 0; c < 256; ++c) {
    float v = bu[b * 256 + c];
    const float* w = fc_w + c * 4;
    z0 += v * w[0]; z1 += v * w[1]; z2 += v * w[2]; z3 += v * w[3];
  }
  for (int c = 0; c < 256; ++c) {
    float v = td[b * 256 + c];
    const float* w = fc_w + (256 + c) * 4;
    z0 += v * w[0]; z1 += v * w[1]; z2 += v * w[2]; z3 += v * w[3];
  }
  float m = fmaxf(fmaxf(z0, z1), fmaxf(z2, z3));
  float l = logf(expf(z0 - m) + expf(z1 - m) + expf(z2 - m) + expf(z3 - m));
  logits[b * 4 + 0] = z0 - m - l;
  logits[b * 4 + 1] = z1 - m - l;
  logits[b * 4 + 2] = z2 - m - l;
  logits[b * 4 + 3] = z3 - m - l;
}

// ------------------------------ host driver ---------------------------------

extern "C" void kernel_launch(void* const* d_in, const int* in_sizes, int n_in,
                              void* d_out, int out_size, void* d_ws, size_t ws_size,
                              hipStream_t stream) {
  (void)in_sizes; (void)n_in; (void)out_size; (void)ws_size;

  const float* x          = (const float*)d_in[0];
  const float* user_state = (const float*)d_in[1];
  // params dict, insertion order:
  const float* td_w1 = (const float*)d_in[2];
  const float* td_b1 = (const float*)d_in[3];
  const float* td_w2 = (const float*)d_in[4];
  const float* td_b2 = (const float*)d_in[5];
  const float* bu_w1 = (const float*)d_in[6];
  const float* bu_b1 = (const float*)d_in[7];
  const float* bu_w2 = (const float*)d_in[8];
  const float* bu_b2 = (const float*)d_in[9];
  const float* wu0_w = (const float*)d_in[10];
  const float* wu0_b = (const float*)d_in[11];
  const float* ws0_b = (const float*)d_in[13];   // ws0_w multiplies zeros
  const float* wd0_b = (const float*)d_in[15];   // wd0_w multiplies zeros
  const float* wu_w  = (const float*)d_in[16];
  const float* wu_b  = (const float*)d_in[17];
  const float* ws_w  = (const float*)d_in[18];
  const float* ws_b  = (const float*)d_in[19];
  const float* wd_w  = (const float*)d_in[20];
  const float* wd_b  = (const float*)d_in[21];
  const float* wx_w  = (const float*)d_in[22];
  const float* wx_b  = (const float*)d_in[23];
  const float* lu_w  = (const float*)d_in[24];
  const float* lu_b  = (const float*)d_in[25];
  const float* ls_w  = (const float*)d_in[26];
  const float* ls_b  = (const float*)d_in[27];
  const float* ld_w  = (const float*)d_in[28];
  const float* ld_b  = (const float*)d_in[29];
  const float* fc_w  = (const float*)d_in[30];
  const float* fc_b  = (const float*)d_in[31];
  const float* raw_alpha = (const float*)d_in[32];
  const float* raw_beta  = (const float*)d_in[33];
  const int* edge_index  = (const int*)d_in[34];
  const int* num_hop     = (const int*)d_in[36];

  const int* src = edge_index;
  const int* dst = edge_index + E_EDGES;

  // ---- workspace layout ----
  char* W = (char*)d_ws;
  size_t off = 0;
  auto take = [&](size_t bytes) -> char* {
    off = (off + 255) & ~(size_t)255;
    char* p = W + off;
    off += bytes;
    return p;
  };
  __bf16* xb      = (__bf16*)take((size_t)N_NODES * F_IN * 2);          // 32 MB
  __bf16* w1T     = (__bf16*)take((size_t)128 * 384 * 2);
  __bf16* w2T     = (__bf16*)take((size_t)128 * 384 * 2);
  float*  dinv_td = (float*) take((size_t)N_NODES * 4);
  float*  dinv_bu = (float*) take((size_t)N_NODES * 4);
  float*  xg      = (float*) take((size_t)B_GRAPH * 256 * 4);
  float*  bout    = (float*) take((size_t)B_GRAPH * 256 * 4);
  float*  tout    = (float*) take((size_t)B_GRAPH * 256 * 4);
  float*  hw      = (float*) take((size_t)N_NODES * 128 * 4);           // 32 MB
  float*  x2      = (float*) take((size_t)N_NODES * 128 * 4);           // 32 MB
  __bf16* cat1    = (__bf16*)take((size_t)N_NODES * 384 * 2);           // 48 MB
  float*  agg2    = (float*)cat1;  // alias: cat1 dead before agg2 is written

  float* outp  = (float*)d_out;
  float* out_u = outp + B_GRAPH * N_CLASS;                 // 512
  float* out_s = out_u + B_GRAPH * MAXHOP;                 // 1792
  float* out_d = out_s + B_GRAPH * MAXHOP;                 // 3072

  // ---- propagation chain (writes xg and U/S/D head outputs) ----
  prop_kernel<<<B_GRAPH, 128, 0, stream>>>(
      user_state, wu0_w, wu0_b, ws0_b, wd0_b, wu_w, wu_b, ws_w, ws_b,
      wd_w, wd_b, wx_w, wx_b, lu_w, lu_b, ls_w, ls_b, ld_w, ld_b,
      raw_alpha, raw_beta, num_hop, xg, out_u, out_s, out_d);

  // ---- degree / normalization ----
  fill_zero_kernel<<<(N_NODES + 255) / 256, 256, 0, stream>>>(dinv_td, N_NODES);
  fill_zero_kernel<<<(N_NODES + 255) / 256, 256, 0, stream>>>(dinv_bu, N_NODES);
  deg_count_kernel<<<(E_EDGES + 255) / 256, 256, 0, stream>>>(src, dst, dinv_td, dinv_bu, E_EDGES);
  deg_to_dinv_kernel<<<(N_NODES + 255) / 256, 256, 0, stream>>>(dinv_td, dinv_bu, N_NODES);

  // ---- x -> bf16 once ----
  cvt_bf16_kernel<<<(N_NODES * F_IN + 255) / 256, 256, 0, stream>>>(
      xb, x, N_NODES * F_IN);

  // ---- two branches: 0 = TD (src->dst), 1 = BU (dst->src) ----
  for (int br = 0; br < 2; ++br) {
    const float* w1 = br ? bu_w1 : td_w1;
    const float* b1 = br ? bu_b1 : td_b1;
    const float* w2 = br ? bu_w2 : td_w2;
    const float* b2 = br ? bu_b2 : td_b2;
    const int* sp   = br ? dst : src;
    const int* dp   = br ? src : dst;
    const float* dv = br ? dinv_bu : dinv_td;
    float* outbr    = br ? bout : tout;

    wcvt_t_kernel<<<(256 * 128 + 255) / 256, 256, 0, stream>>>(w1T, w1, 256);
    wcvt_t_kernel<<<(384 * 128 + 255) / 256, 256, 0, stream>>>(w2T, w2, 384);

    // layer 1: hw = x @ w1
    gemm_bf16_wmma_kernel<<<N_NODES / 128, 256, 0, stream>>>(hw, xb, w1T, 256);
    // x2 = self-loop + edge aggregation + b1
    agg_init_kernel<<<(N_NODES * 128) / 256, 256, 0, stream>>>(
        x2, hw, dv, b1, N_NODES * 128);
    edge_scatter_kernel<<<(E_EDGES * 32) / 256, 256, 0, stream>>>(
        x2, hw, sp, dp, dv, E_EDGES);
    // cat1 = relu(concat(x2, x[root])) in bf16
    cat_relu_kernel<<<(N_NODES * 384 + 255) / 256, 256, 0, stream>>>(
        cat1, x2, x, N_NODES * 384);
    // layer 2: hw = cat1 @ w2
    gemm_bf16_wmma_kernel<<<N_NODES / 128, 256, 0, stream>>>(hw, cat1, w2T, 384);
    // agg2 = self-loop + edge aggregation + b2 (aliases cat1, now dead)
    agg_init_kernel<<<(N_NODES * 128) / 256, 256, 0, stream>>>(
        agg2, hw, dv, b2, N_NODES * 128);
    edge_scatter_kernel<<<(E_EDGES * 32) / 256, 256, 0, stream>>>(
        agg2, hw, sp, dp, dv, E_EDGES);
    // pooled mean(relu(agg2)) ++ x2[root], plus xg
    pool_kernel<<<B_GRAPH, 128, 0, stream>>>(outbr, agg2, x2, xg);
  }

  // ---- logits = concat(bu, td) @ fc_w + fc_b; log_softmax ----
  final_kernel<<<1, 128, 0, stream>>>(outp, bout, tout, fc_w, fc_b);
}
